// PoP_CSNet_71090298683669
// MI455X (gfx1250) — compile-verified, matrix operand order checked by reading, not run on hardware
//
#include <hip/hip_runtime.h>

#define HH 256
#define WW 256
#define NB 4

typedef __attribute__((ext_vector_type(16))) _Float16 v16h;
typedef __attribute__((ext_vector_type(8)))  float    v8f;

// ---------------------------------------------------------------------------
// Sampling conv: out[b,m,i,j] = sum_{32x32 patch} w[m,r,s] * img[b, 32i+r, 32j+s]
// ---------------------------------------------------------------------------
__global__ __launch_bounds__(256) void samp_kernel(const float* __restrict__ img,
                                                   const float* __restrict__ w,
                                                   float* __restrict__ out) {
  __shared__ float patch[1024];
  int blk = blockIdx.x;             // b*64 + i*8 + j
  int j = blk & 7, i = (blk >> 3) & 7, b = blk >> 6;
  int tid = threadIdx.x;
  for (int e = 0; e < 4; ++e) {
    int t = tid + e * 256;
    int r = t >> 5, s = t & 31;
    patch[t] = img[((b * HH) + i * 32 + r) * WW + j * 32 + s];
  }
  __syncthreads();
  if (tid < 102) {
    float acc = 0.f;
    for (int t = 0; t < 1024; ++t) acc += w[tid * 1024 + t] * patch[t];
    out[((b * 102 + tid) * 8 + i) * 8 + j] = acc;
  }
}

// z = y - t
__global__ void initz_kernel(float* __restrict__ z, const float* __restrict__ y,
                             const float* __restrict__ t, int n) {
  int i = blockIdx.x * blockDim.x + threadIdx.x;
  if (i < n) z[i] = y[i] - t[i];
}

// z += lam * (y - t)
__global__ void updz_kernel(float* __restrict__ z, const float* __restrict__ y,
                            const float* __restrict__ t, const float* __restrict__ lam, int n) {
  int i = blockIdx.x * blockDim.x + threadIdx.x;
  if (i < n) z[i] += lam[0] * (y[i] - t[i]);
}

__global__ void zero_kernel(float* __restrict__ p, int n) {
  int i = blockIdx.x * blockDim.x + threadIdx.x;
  if (i < n) p[i] = 0.f;
}

// ---------------------------------------------------------------------------
// Fused: res = block_to_image(conv1x1(z, w_ups)); then
//   mode 0: cur = res
//   mode 1: cur = x1 + lamx*res; res_acc += lamres*res
// ---------------------------------------------------------------------------
__global__ __launch_bounds__(256) void ups_image_kernel(const float* __restrict__ z,
                                                        const float* __restrict__ wups,
                                                        const float* __restrict__ x1,
                                                        const float* __restrict__ lamx,
                                                        const float* __restrict__ lamres,
                                                        float* __restrict__ cur,
                                                        float* __restrict__ resacc,
                                                        int mode) {
  int p = blockIdx.x * 256 + threadIdx.x;          // < NB*HH*WW
  int X = p & 255, Y = (p >> 8) & 255, b = p >> 16;
  int i = Y >> 5, r = Y & 31, j = X >> 5, s = X & 31;
  int c = r * 32 + s;
  const float* wr = wups + c * 102;
  const float* zr = z + ((b * 102) * 8 + i) * 8 + j;
  float acc = 0.f;
  for (int m = 0; m < 102; ++m) acc += wr[m] * zr[m * 64];
  if (mode == 0) {
    cur[p] = acc;
  } else {
    cur[p] = x1[p] + lamx[0] * acc;
    resacc[p] += lamres[0] * acc;
  }
}

// ---------------------------------------------------------------------------
// conv RxR from 1 input channel + bias + PReLU(scalar alpha).
// ---------------------------------------------------------------------------
template <int COUT, int R>
__global__ __launch_bounds__(256) void conv_from1_kernel(const float* __restrict__ img,
                                                         const float* __restrict__ w,
                                                         const float* __restrict__ bias,
                                                         const float* __restrict__ alpha,
                                                         float* __restrict__ out,
                                                         int octotal, int ocbase) {
  constexpr int PAD = R / 2;
  constexpr int TS = 16 + R - 1;
  __shared__ float s_in[TS * TS];
  __shared__ float s_w[COUT * R * R];
  int blk = blockIdx.x;               // b*256 + by*16 + bx
  int bx = blk & 15, by = (blk >> 4) & 15, b = blk >> 8;
  int x0 = bx * 16 - PAD, y0 = by * 16 - PAD;
  int tid = threadIdx.x;
  for (int t = tid; t < TS * TS; t += 256) {
    int yy = y0 + t / TS, xx = x0 + t % TS;
    float v = 0.f;
    if (yy >= 0 && yy < HH && xx >= 0 && xx < WW) v = img[(b * HH + yy) * WW + xx];
    s_in[t] = v;
  }
  for (int t = tid; t < COUT * R * R; t += 256) s_w[t] = w[t];
  __syncthreads();
  int ty = tid >> 4, tx = tid & 15;
  float acc[COUT];
#pragma unroll
  for (int c = 0; c < COUT; ++c) acc[c] = 0.f;
  for (int dy = 0; dy < R; ++dy)
    for (int dx = 0; dx < R; ++dx) {
      float iv = s_in[(ty + dy) * TS + tx + dx];
#pragma unroll
      for (int c = 0; c < COUT; ++c) acc[c] += s_w[c * R * R + dy * R + dx] * iv;
    }
  float a = alpha[0];
  int y = by * 16 + ty, x = bx * 16 + tx;
#pragma unroll
  for (int c = 0; c < COUT; ++c) {
    float v = acc[c] + bias[c];
    v = v > 0.f ? v : a * v;
    out[((b * octotal + ocbase + c) * HH + y) * WW + x] = v;
  }
}

// ---------------------------------------------------------------------------
// conv3x3 to a single output channel (+bias), with optional residual modes:
//   mode 0: out = conv        mode 1: out = conv + add        mode 2: out = add - conv
// ---------------------------------------------------------------------------
__global__ __launch_bounds__(256) void conv_to1_kernel(const float* __restrict__ in,
                                                       int ctotal, int cbase, int cin,
                                                       const float* __restrict__ w,
                                                       const float* __restrict__ bias,
                                                       const float* __restrict__ add,
                                                       int mode, float* __restrict__ out) {
  int p = blockIdx.x * 256 + threadIdx.x;
  int x = p & 255, y = (p >> 8) & 255, b = p >> 16;
  float acc = bias[0];
  for (int ci = 0; ci < cin; ++ci) {
    const float* base = in + ((size_t)(b * ctotal + cbase + ci) * HH) * WW;
    const float* wr = w + ci * 9;
#pragma unroll
    for (int dy = 0; dy < 3; ++dy) {
      int yy = y + dy - 1;
      if (yy < 0 || yy >= HH) continue;
#pragma unroll
      for (int dx = 0; dx < 3; ++dx) {
        int xx = x + dx - 1;
        if (xx < 0 || xx >= WW) continue;
        acc += wr[dy * 3 + dx] * base[yy * WW + xx];
      }
    }
  }
  float v;
  if (mode == 0) v = acc;
  else if (mode == 1) v = acc + add[p];
  else v = add[p] - acc;
  out[p] = v;
}

// ---------------------------------------------------------------------------
// WMMA implicit-GEMM conv, fp32 NCHW tensors, f16 multiply / f32 accumulate.
// Compile-time MT (m-tiles, cout = 16*MT) and R (1 or 3):
//  - static WMMA loop (no conditional accumulator updates)
//  - R==1 fast staging path: no tap decomposition, no spatial bounds checks
//  - BN pre-affine (g/sqrt(1+1e-5), b) cached in LDS once per block
//   Post-ops (optional): +bias[co], +add tensor, PReLU(post_a), soft-threshold(thr)
// M = 16*MT, N = 128 pixels (x-run of one row), K = cin*R*R padded to 32.
// LDS fragments pre-swizzled to the ISA 16-bit A/B per-lane layout so each lane
// loads its v16h with two ds_load_b128.
// ---------------------------------------------------------------------------
template <int MT, int R>
__global__ __launch_bounds__(256) void wmma_conv_kernel(
    const float* __restrict__ in, int ictotal, int icbase, int cin,
    const float* __restrict__ wgt,
    const float* __restrict__ bias,
    const float* __restrict__ pre_g, const float* __restrict__ pre_b,
    const float* __restrict__ pre_a,
    const float* __restrict__ post_a,
    const float* __restrict__ add, int add_ctotal,
    const float* __restrict__ thr,
    float* __restrict__ out, int octotal, int ocbase) {
  constexpr int RR = R * R;
  constexpr int PAD = R >> 1;
  constexpr int COUT = MT * 16;
  __shared__ v16h Afrag[MT * 32];  // [mtile][lane]
  __shared__ v16h Bfrag[8 * 32];   // [ntile][lane]
  __shared__ float s_scale[128];
  __shared__ float s_shift[128];
  _Float16* Ah = (_Float16*)Afrag;
  _Float16* Bh = (_Float16*)Bfrag;

  int blk = blockIdx.x;            // b*(H*2) + y*2 + xseg
  int xseg = blk & 1;
  int y = (blk >> 1) & 255;
  int b = blk >> 9;
  int xbase = xseg * 128;
  int tid = threadIdx.x;
  int lane = tid & 31;
  int wid = tid >> 5;              // n-tile index (0..7)

  int Ktot = cin * RR;
  bool has_pre = (pre_g != nullptr);
  float pa = pre_a ? pre_a[0] : 0.f;
  const float BNINV = 0.9999950000374997f;  // 1/sqrt(1+1e-5)

  if (has_pre) {
    for (int t = tid; t < cin; t += 256) {
      s_scale[t] = pre_g[t] * BNINV;
      s_shift[t] = pre_b[t];
    }
  }
  __syncthreads();

  v8f zf = {0.f, 0.f, 0.f, 0.f, 0.f, 0.f, 0.f, 0.f};
  v8f acc[MT];
#pragma unroll
  for (int i = 0; i < MT; ++i) acc[i] = zf;

  for (int k0 = 0; k0 < Ktot; k0 += 32) {
    // ---- stage B: 32(K) x 128(N), pre-op applied, swizzled into fragments ----
#pragma unroll
    for (int e = 0; e < 16; ++e) {
      int idx = e * 256 + tid;
      int kk = idx >> 7;           // 0..31
      int n  = idx & 127;          // pixel within tile (x-contiguous per kk)
      int k = k0 + kk;
      float v = 0.f;
      if constexpr (R == 1) {
        // 1x1: ci == k, pixel always in range
        if (k < Ktot) {
          v = in[(((size_t)b * ictotal + icbase + k) * HH + y) * WW + xbase + n];
          if (has_pre) {
            v = v * s_scale[k] + s_shift[k];
            v = v > 0.f ? v : pa * v;
          }
        }
      } else {
        if (k < Ktot) {
          int ci = k / RR;         // RR constexpr -> strength-reduced
          int tap = k - ci * RR;
          int dy = tap / R - PAD;
          int dx = tap - (tap / R) * R - PAD;
          int yy = y + dy;
          int xx = xbase + n + dx;
          if (yy >= 0 && yy < HH && xx >= 0 && xx < WW) {
            v = in[(((size_t)b * ictotal + icbase + ci) * HH + yy) * WW + xx];
            if (has_pre) {
              v = v * s_scale[ci] + s_shift[ci];
              v = v > 0.f ? v : pa * v;
            }
          }
        }
      }
      int ntile = n >> 4, nn = n & 15;
      int g  = (kk >> 3) & 1;
      int hh = (kk & 7) + ((kk >> 4) << 3);
      Bh[((ntile * 32) + nn + (g << 4)) * 16 + hh] = (_Float16)v;
    }
    // ---- stage A: COUT(M) x 32(K) weights, swizzled into fragments ----
#pragma unroll
    for (int e = 0; e < (COUT * 32 + 255) / 256; ++e) {
      int idx = e * 256 + tid;
      if (idx < COUT * 32) {
        int m = idx >> 5;
        int kk = idx & 31;
        int k = k0 + kk;
        float v = (k < Ktot) ? wgt[m * Ktot + k] : 0.f;
        int mtile = m >> 4, mm = m & 15;
        int g  = (kk >> 3) & 1;
        int hh = (kk & 7) + ((kk >> 4) << 3);
        Ah[((mtile * 32) + mm + (g << 4)) * 16 + hh] = (_Float16)v;
      }
    }
    __syncthreads();
    v16h bfr = Bfrag[wid * 32 + lane];
#pragma unroll
    for (int mt = 0; mt < MT; ++mt) {
      v16h afr = Afrag[mt * 32 + lane];
      acc[mt] = __builtin_amdgcn_wmma_f32_16x16x32_f16(
          false, afr, false, bfr, (short)0, acc[mt], false, false);
    }
    __syncthreads();
  }

  // ---- epilogue: decode C/D layout, apply post-ops, store fp32 ----
  float poa = post_a ? post_a[0] : 0.f;
  float th = thr ? thr[0] : 0.f;
  int g = lane >> 4;
  int nn = lane & 15;
  int x = xbase + wid * 16 + nn;
#pragma unroll
  for (int mt = 0; mt < MT; ++mt) {
#pragma unroll
    for (int r = 0; r < 8; ++r) {
      int co = mt * 16 + r + (g << 3);
      float v = acc[mt][r];
      if (bias) v += bias[co];
      if (add) v += add[(((size_t)b * add_ctotal + co) * HH + y) * WW + x];
      if (post_a) v = v > 0.f ? v : poa * v;
      if (thr) {
        float av = fabsf(v) - th;
        av = av > 0.f ? av : 0.f;
        v = copysignf(av, v);
      }
      out[(((size_t)b * octotal + ocbase + co) * HH + y) * WW + x] = v;
    }
  }
}

// ---------------------------------------------------------------------------
// Host-side orchestration
// ---------------------------------------------------------------------------
extern "C" void kernel_launch(void* const* d_in, const int* in_sizes, int n_in,
                              void* d_out, int out_size, void* d_ws, size_t ws_size,
                              hipStream_t stream) {
  (void)in_sizes; (void)n_in; (void)out_size; (void)ws_size;

  const float* x      = (const float*)d_in[0];
  const float* w_samp = (const float*)d_in[1];
  const float* w_ups  = (const float*)d_in[2];
  const float* w_up   = (const float*)d_in[3];
  const float* b_up   = (const float*)d_in[4];
  const float* a_up   = (const float*)d_in[5];
  const float* w_down = (const float*)d_in[6];
  const float* b_down = (const float*)d_in[7];
  const float* w_same = (const float*)d_in[8];
  const float* b_same = (const float*)d_in[9];
  const float* a_same = (const float*)d_in[10];
  // dense layers: indices 11 + li*8 : {g1,b1,a1,w1,g2,b2,a2,w2}
  const float* tg     = (const float*)d_in[43];
  const float* tb     = (const float*)d_in[44];
  const float* ta     = (const float*)d_in[45];
  const float* w_tr   = (const float*)d_in[46];
  const float* b_tr   = (const float*)d_in[47];
  const float* w_d1   = (const float*)d_in[48];
  const float* b_d1   = (const float*)d_in[49];
  const float* a_d1   = (const float*)d_in[50];
  const float* w_d2   = (const float*)d_in[51];
  const float* b_d2   = (const float*)d_in[52];
  const float* a_d2   = (const float*)d_in[53];
  const float* w_d3   = (const float*)d_in[54];
  const float* b_d3   = (const float*)d_in[55];
  const float* a_d3   = (const float*)d_in[56];
  const float* w_d4   = (const float*)d_in[57];
  const float* b_d4   = (const float*)d_in[58];
  const float* lam_x  = (const float*)d_in[59];
  const float* lam_z  = (const float*)d_in[60];
  const float* thr    = (const float*)d_in[61];
  const float* lam_res= (const float*)d_in[62];

  char* wp = (char*)d_ws;
  float* dense = (float*)wp; wp += (size_t)NB * 128 * HH * WW * 4;  // concat buffer
  float* t64   = (float*)wp; wp += (size_t)NB * 64  * HH * WW * 4;  // temp 64ch
  float* imgA  = (float*)wp; wp += (size_t)NB * HH * WW * 4;        // cur
  float* imgB  = (float*)wp; wp += (size_t)NB * HH * WW * 4;        // res_acc
  float* imgC  = (float*)wp; wp += (size_t)NB * HH * WW * 4;        // x1 / f
  float* ybuf  = (float*)wp; wp += 131072;
  float* zbuf  = (float*)wp; wp += 131072;
  float* tmpm  = (float*)wp; wp += 131072;
  float* t32a  = t64;
  float* t32b  = t64 + (size_t)NB * 32 * HH * WW;

  const int nm = NB * 102 * 64;          // measurement elements
  const int npix = NB * HH * WW;

  // y = samp(x); cur = block_to_image(conv(y, w_ups))
  samp_kernel<<<256, 256, 0, stream>>>(x, w_samp, ybuf);
  ups_image_kernel<<<1024, 256, 0, stream>>>(ybuf, w_ups, nullptr, nullptr, nullptr,
                                             imgA, nullptr, 0);
  // z = y - samp(cur); res_acc = 0
  samp_kernel<<<256, 256, 0, stream>>>(imgA, w_samp, tmpm);
  initz_kernel<<<(nm + 255) / 256, 256, 0, stream>>>(zbuf, ybuf, tmpm, nm);
  zero_kernel<<<(npix + 255) / 256, 256, 0, stream>>>(imgB, npix);

  for (int k = 0; k < 9; ++k) {
    // up = prelu(conv7x7(cur)) -> dense channels [0:64)
    conv_from1_kernel<64, 7><<<1024, 256, 0, stream>>>(imgA, w_up, b_up, a_up,
                                                       dense, 128, 0);
    for (int li = 0; li < 4; ++li) {
      int cin = 64 + 16 * li;
      const float* g1 = (const float*)d_in[11 + li * 8 + 0];
      const float* b1 = (const float*)d_in[11 + li * 8 + 1];
      const float* a1 = (const float*)d_in[11 + li * 8 + 2];
      const float* w1 = (const float*)d_in[11 + li * 8 + 3];
      const float* g2 = (const float*)d_in[11 + li * 8 + 4];
      const float* b2 = (const float*)d_in[11 + li * 8 + 5];
      const float* a2 = (const float*)d_in[11 + li * 8 + 6];
      const float* w2 = (const float*)d_in[11 + li * 8 + 7];
      // 1x1 bottleneck: bn+prelu fused as pre-op
      wmma_conv_kernel<4, 1><<<2048, 256, 0, stream>>>(dense, 128, 0, cin, w1,
          nullptr, g1, b1, a1, nullptr, nullptr, 0, nullptr, t64, 64, 0);
      // 3x3 growth conv -> concat slot
      wmma_conv_kernel<1, 3><<<2048, 256, 0, stream>>>(t64, 64, 0, 64, w2,
          nullptr, g2, b2, a2, nullptr, nullptr, 0, nullptr, dense, 128, 64 + 16 * li);
    }
    // transition 128->64 (+b_tr) + up residual + soft-threshold -> xs in t64
    wmma_conv_kernel<4, 1><<<2048, 256, 0, stream>>>(dense, 128, 0, 128, w_tr,
        b_tr, tg, tb, ta, nullptr, dense, 128, thr + k, t64, 64, 0);
    // x1 = conv3x3(xs, w_down)
    conv_to1_kernel<<<1024, 256, 0, stream>>>(t64, 64, 0, 64, w_down, b_down,
                                              nullptr, 0, imgC);
    // z += lam_z * (y - samp(x1))
    samp_kernel<<<256, 256, 0, stream>>>(imgC, w_samp, tmpm);
    updz_kernel<<<(nm + 255) / 256, 256, 0, stream>>>(zbuf, ybuf, tmpm, lam_z + k, nm);
    // res = b2i(conv(z, w_ups)); cur = x1 + lam_x*res; res_acc += lam_res*res
    ups_image_kernel<<<1024, 256, 0, stream>>>(zbuf, w_ups, imgC, lam_x + k,
                                               lam_res + k, imgA, imgB, 1);
  }

  // f1 = prelu(conv7x7(cur)); f2 = prelu(conv3x3(f1, w_same)); f = conv3x3(f2) + res_acc
  conv_from1_kernel<64, 7><<<1024, 256, 0, stream>>>(imgA, w_up, b_up, a_up,
                                                     dense, 128, 0);
  wmma_conv_kernel<4, 3><<<2048, 256, 0, stream>>>(dense, 128, 0, 64, w_same,
      b_same, nullptr, nullptr, nullptr, a_same, nullptr, 0, nullptr, t64, 64, 0);
  conv_to1_kernel<<<1024, 256, 0, stream>>>(t64, 64, 0, 64, w_down, b_down,
                                            imgB, 1, imgC);   // f

  // deblock: out = f - conv(prelu-chain(f))
  conv_from1_kernel<32, 3><<<1024, 256, 0, stream>>>(imgC, w_d1, b_d1, a_d1,
                                                     t32a, 32, 0);
  wmma_conv_kernel<2, 3><<<2048, 256, 0, stream>>>(t32a, 32, 0, 32, w_d2,
      b_d2, nullptr, nullptr, nullptr, a_d2, nullptr, 0, nullptr, t32b, 32, 0);
  wmma_conv_kernel<2, 3><<<2048, 256, 0, stream>>>(t32b, 32, 0, 32, w_d3,
      b_d3, nullptr, nullptr, nullptr, a_d3, nullptr, 0, nullptr, t32a, 32, 0);
  conv_to1_kernel<<<1024, 256, 0, stream>>>(t32a, 32, 0, 32, w_d4, b_d4,
                                            imgC, 2, (float*)d_out);
}
